// Detector_80685255622816
// MI455X (gfx1250) — compile-verified, matrix operand order checked by reading, not run on hardware
//
#include <hip/hip_runtime.h>
#include <hip/hip_bf16.h>

typedef __bf16 v16bf __attribute__((ext_vector_type(16)));
typedef float  v8f   __attribute__((ext_vector_type(8)));

// ---------------------------------------------------------------------------
// Weight packing: OIHW fp32 [Co][Ksrc] -> bf16 WMMA 16-bit fragment order,
// K padded with zeros to Kpad (multiple of 32). Within each 32-wide K block,
// k_inner = g*16 + khalf*8 + w  ->  packed position p = khalf*16 + g*8 + w,
// so each lane's 16 halfs are one 32B contiguous run.
// ---------------------------------------------------------------------------
__global__ __launch_bounds__(256)
void pack_weights(const float* __restrict__ src, __bf16* __restrict__ dst,
                  int Co, int Ksrc, int Kpad)
{
    int i = blockIdx.x * 256 + threadIdx.x;
    int n = Co * Kpad;
    if (i >= n) return;
    int co = i / Kpad;
    int k  = i - co * Kpad;
    float v = (k < Ksrc) ? src[(size_t)co * Ksrc + k] : 0.0f;
    int kb = k >> 5;
    int ki = k & 31;
    int g = (ki >> 4) & 1, kh = (ki >> 3) & 1, w = ki & 7;
    dst[((size_t)kb * Co + co) * 32 + kh * 16 + g * 8 + w] = (__bf16)v;
}

// Fold BN into per-channel scale/bias:  y = x*scale + bias
__global__ void bn_prep(const float* __restrict__ g, const float* __restrict__ b,
                        const float* __restrict__ m, const float* __restrict__ v,
                        float* __restrict__ scale, float* __restrict__ bias, int C)
{
    int c = blockIdx.x * 256 + threadIdx.x;
    if (c < C) {
        float inv = g[c] * rsqrtf(v[c] + 1e-5f);
        scale[c] = inv;
        bias[c]  = b[c] - m[c] * inv;
    }
}

__global__ void bias_prep(const float* __restrict__ b,
                          float* __restrict__ scale, float* __restrict__ bias, int C)
{
    int c = blockIdx.x * 256 + threadIdx.x;
    if (c < C) { scale[c] = 1.0f; bias[c] = b[c]; }
}

__global__ __launch_bounds__(256)
void f32_to_bf16(const float* __restrict__ src, __bf16* __restrict__ dst, int n)
{
    int i = blockIdx.x * 256 + threadIdx.x;
    if (i < n) dst[i] = (__bf16)src[i];
}

// ---------------------------------------------------------------------------
// Implicit-GEMM WMMA conv, bf16 activations in/out, fp32 accumulate.
//   GEMM: M = Co (A = packed weights), N = 8*Ho*Wo pixels (B = im2col LDS),
//         K = Ci*KS*KS padded to Kpad (mult of 32).
//   8 waves = WM x WN of 32x32 per-wave tiles (2x2 v_wmma_f32_16x16x32_bf16).
//   Square spatial maps only; Wo = 1<<woShift.
//   Double-buffered LDS tile: one barrier per K-step, fill(kb+1) overlaps
//   the WMMAs of kb. Pixel decode hoisted out of the K loop. The im2col fill
//   is branchless (clamped load + cndmask-select of zero) so it fully
//   unrolls with uniform control flow.
// ---------------------------------------------------------------------------
template <int CO_TILE, int KS>
__global__ __launch_bounds__(256)
void wmma_conv(const __bf16* __restrict__ in, const __bf16* __restrict__ wp,
               const float* __restrict__ scale, const float* __restrict__ bias,
               const __bf16* __restrict__ residual, __bf16* __restrict__ out,
               int Ci, int Hi, int Wi, int Co, int woShift,
               int stride, int pad, int relu, int K, int Kpad)
{
    constexpr int WM = CO_TILE / 32;        // waves along M (Co)
    constexpr int WN = 8 / WM;              // waves along N (pixels)
    constexpr int N_TILE = WN * 32;         // 128 or 256
    constexpr int KI_STEP = 256 / N_TILE;   // 2 or 1
    constexpr int KSQ = KS * KS;

    __shared__ __bf16 ldsB[2][N_TILE * 40]; // 40-half row stride, 16B-aligned rows

    const int t     = threadIdx.x;
    const int lane  = t & 31;
    const int n16   = lane & 15;
    const int khalf = (lane >> 4) & 1;
    const int wave  = t >> 5;
    const int wm    = wave % WM;
    const int wn    = wave / WM;

    const int Wo   = 1 << woShift;
    const int HoWo = 1 << (2 * woShift);
    const int pixBase = blockIdx.x * N_TILE;
    const int coBase  = blockIdx.y * CO_TILE;

    // ---- loop-invariant im2col coordinates for this thread ---------------
    const int jf     = t % N_TILE;          // fixed pixel within tile
    const int kiBase = t / N_TILE;          // 0..KI_STEP-1
    int pixF = pixBase + jf;
    int bF   = pixF >> (2 * woShift);
    int remF = pixF & (HoWo - 1);
    int iy0  = (remF >> woShift) * stride - pad;
    int ix0  = (remF & (Wo - 1)) * stride - pad;
    const __bf16* inB = in + (size_t)bF * Ci * Hi * Wi;
    const int ldsRow = jf * 40;

    auto fill = [&](int buf, int kb) {
#pragma unroll
        for (int ki = kiBase; ki < 32; ki += KI_STEP) {
            int k = (kb << 5) + ki;
            int ci = k / KSQ; int r = k - ci * KSQ;  // constant divisors
            int ky = r / KS;  int kx = r - ky * KS;
            int iy = iy0 + ky, ix = ix0 + kx;
            // branchless: clamp address into range, select zero afterwards
            bool ok = (k < K) & (iy >= 0) & (iy < Hi) & (ix >= 0) & (ix < Wi);
            int iyc = iy < 0 ? 0 : (iy >= Hi ? Hi - 1 : iy);
            int ixc = ix < 0 ? 0 : (ix >= Wi ? Wi - 1 : ix);
            int cic = (k < K) ? ci : 0;
            __bf16 v = inB[((size_t)cic * Hi + iyc) * Wi + ixc];
            v = ok ? v : (__bf16)0.0f;
            ldsB[buf][ldsRow + ((ki >> 3) & 1) * 16 + ((ki >> 4) & 1) * 8 + (ki & 7)] = v;
        }
    };

    v8f acc[2][2] = {};
    const int Kb = Kpad >> 5;

    fill(0, 0);
    __syncthreads();

    for (int kb = 0; kb < Kb; ++kb) {
        const int cur = kb & 1;

        // A fragments: 32B contiguous per lane from packed weights (global)
        v16bf afr[2];
#pragma unroll
        for (int mt = 0; mt < 2; ++mt) {
            int co = coBase + wm * 32 + mt * 16 + n16;
            const uint4* p = (const uint4*)(wp + ((size_t)kb * Co + co) * 32 + khalf * 16);
            uint4* d = (uint4*)&afr[mt];
            d[0] = p[0]; d[1] = p[1];
        }
        // B fragments: 32B contiguous per lane from LDS
        v16bf bfr[2];
#pragma unroll
        for (int nt = 0; nt < 2; ++nt) {
            int j = wn * 32 + nt * 16 + n16;
            const uint4* p = (const uint4*)(&ldsB[cur][j * 40 + khalf * 16]);
            uint4* d = (uint4*)&bfr[nt];
            d[0] = p[0]; d[1] = p[1];
        }

        if (kb + 1 < Kb) fill(1 - cur, kb + 1);  // overlaps with WMMAs below

#pragma unroll
        for (int mt = 0; mt < 2; ++mt)
#pragma unroll
            for (int nt = 0; nt < 2; ++nt)
                acc[mt][nt] = __builtin_amdgcn_wmma_f32_16x16x32_bf16(
                    false, afr[mt], false, bfr[nt], (short)0, acc[mt][nt],
                    false, false);

        __syncthreads();
    }

    // ---- epilogue: D lane holds N = lane%16 (pixel), M = el + 8*khalf ----
#pragma unroll
    for (int nt = 0; nt < 2; ++nt) {
        int pix = pixBase + wn * 32 + nt * 16 + n16;
        int b   = pix >> (2 * woShift);
        int rem = pix & (HoWo - 1);
        int oy  = rem >> woShift;
        int ox  = rem & (Wo - 1);
#pragma unroll
        for (int mt = 0; mt < 2; ++mt) {
#pragma unroll
            for (int el = 0; el < 8; ++el) {
                int co = coBase + wm * 32 + mt * 16 + el + 8 * khalf;
                float val = acc[mt][nt][el] * scale[co] + bias[co];
                size_t oi = (((size_t)b * Co + co) << (2 * woShift)) + (oy << woShift) + ox;
                if (residual) val += (float)residual[oi];
                if (relu) val = fmaxf(val, 0.0f);
                out[oi] = (__bf16)val;
            }
        }
    }
}

// maxpool 3x3 s2 pad1 (bf16): [8,64,256,256] -> [8,64,128,128]
__global__ __launch_bounds__(256)
void maxpool3x3s2(const __bf16* __restrict__ in, __bf16* __restrict__ out)
{
    size_t i = (size_t)blockIdx.x * 256 + threadIdx.x;   // 8*64*128*128
    int ox = i & 127;
    int oy = (i >> 7) & 127;
    int c  = (int)(i >> 14);                              // b*64+co
    float m = -3.402823466e38f;
    for (int ky = 0; ky < 3; ++ky) {
        int iy = oy * 2 + ky - 1;
        if (iy < 0 || iy >= 256) continue;
        for (int kx = 0; kx < 3; ++kx) {
            int ix = ox * 2 + kx - 1;
            if (ix < 0 || ix >= 256) continue;
            m = fmaxf(m, (float)in[((size_t)c * 256 + iy) * 256 + ix]);
        }
    }
    out[i] = (__bf16)m;
}

// head conv2: 1x1, 32 -> 3, with bias; bf16 feat -> fp32 heatmap
__global__ __launch_bounds__(256)
void conv1x1_head(const __bf16* __restrict__ feat, const float* __restrict__ w,
                  const float* __restrict__ b2, float* __restrict__ hm3)
{
    int i = blockIdx.x * 256 + threadIdx.x;    // 8*3*1024
    if (i >= 8 * 3 * 1024) return;
    int p  = i & 1023;
    int co = (i >> 10) % 3;
    int b  = (i >> 10) / 3;
    float acc = b2[co];
    for (int c = 0; c < 32; ++c)
        acc += (float)feat[((size_t)b * 32 + c) * 1024 + p] * w[co * 32 + c];
    hm3[i] = acc;
}

// spatial_softmax (15x15 VALID window, edge-padded) + 5x5 NMS, per batch
__global__ __launch_bounds__(256)
void softmax_nms(const float* __restrict__ hm3, float* __restrict__ nmsed)
{
    int b = blockIdx.x, t = threadIdx.x;
    __shared__ float ex[1024];
    __shared__ float sc[1024];
    __shared__ float s18[18 * 18];
    __shared__ float red[256];
    const float* h = hm3 + (size_t)b * 3 * 1024;

    float m = -3.402823466e38f;
    for (int i = t; i < 1024; i += 256) m = fmaxf(m, h[i]);
    red[t] = m; __syncthreads();
    for (int s = 128; s > 0; s >>= 1) {
        if (t < s) red[t] = fmaxf(red[t], red[t + s]);
        __syncthreads();
    }
    m = red[0];
    for (int i = t; i < 1024; i += 256) ex[i] = expf(10.0f * (h[i] - m));
    __syncthreads();
    for (int o = t; o < 324; o += 256) {
        int oy = o / 18, ox = o % 18;
        float s = 0.0f;
        for (int ky = 0; ky < 15; ++ky)
            for (int kx = 0; kx < 15; ++kx)
                s += ex[(oy + ky) * 32 + ox + kx];
        s18[o] = s;
    }
    __syncthreads();
    for (int i = t; i < 1024; i += 256) {
        int y = i >> 5, x = i & 31;
        int sy = y - 7; sy = sy < 0 ? 0 : (sy > 17 ? 17 : sy);
        int sx = x - 7; sx = sx < 0 ? 0 : (sx > 17 ? 17 : sx);
        sc[i] = ex[i] / (s18[sy * 18 + sx] + 1e-6f);
    }
    __syncthreads();
    for (int i = t; i < 1024; i += 256) {
        int y = i >> 5, x = i & 31;
        float mx = -3.402823466e38f;
        for (int dy = -2; dy <= 2; ++dy) {
            int yy = y + dy; if (yy < 0 || yy > 31) continue;
            for (int dx = -2; dx <= 2; ++dx) {
                int xx = x + dx; if (xx < 0 || xx > 31) continue;
                mx = fmaxf(mx, sc[yy * 32 + xx]);
            }
        }
        float v = sc[i];
        nmsed[(size_t)b * 1024 + i] = (v >= mx) ? v : 0.0f;
    }
}

// stable top-512 of 1024 (jax.lax.top_k tie-break: lower index first) + gather
__global__ __launch_bounds__(256)
void topk_gather(const float* __restrict__ nmsed, const float* __restrict__ hm3,
                 float* __restrict__ out)
{
    int b = blockIdx.x, t = threadIdx.x;
    __shared__ float v[1024];
    for (int i = t; i < 1024; i += 256) v[i] = nmsed[(size_t)b * 1024 + i];
    __syncthreads();
    for (int i = t; i < 1024; i += 256) {
        float vi = v[i];
        int rank = 0;
        for (int j = 0; j < 1024; ++j) {
            float vj = v[j];
            rank += (vj > vi) || (vj == vi && j < i);
        }
        if (rank < 512) {
            int x = i & 31, y = i >> 5;
            float rx = fmaxf(hm3[(size_t)b * 3072 + 1024 + i], 0.0f);
            float ry = fmaxf(hm3[(size_t)b * 3072 + 2048 + i], 0.0f);
            float* o = out + ((size_t)b * 512 + rank) * 4;
            o[0] = rx + (float)x;
            o[1] = ry + (float)y;
            o[2] = 32.0f;   // ANCHOR = 512/16
            o[3] = 32.0f;
        }
    }
}

// ---------------------------------------------------------------------------
// Host orchestration
// ---------------------------------------------------------------------------
extern "C" void kernel_launch(void* const* d_in, const int* in_sizes, int n_in,
                              void* d_out, int out_size, void* d_ws, size_t ws_size,
                              hipStream_t stream)
{
    (void)in_sizes; (void)n_in; (void)out_size; (void)ws_size;

    // --- JAX pytree leaf order (sorted dict keys at every level) ----------
    struct BN { int b, g, m, v; };
    struct Blk { BN bn1, bn2, bnd; int w1, w2, wd; bool ds; };
    const int strides[13] = {1,1,1,2,1,1,1,2,1,1,1,1,1};
    const int chO[13] = {64,64,64,128,128,128,128,256,256,256,256,256,256};
    const int chI[13] = {64,64,64, 64,128,128,128,128,256,256,256,256,256};
    Blk blk[13];
    int idx = 1;
    for (int i = 0; i < 13; ++i) {
        bool ds = (strides[i] != 1) || (chI[i] != chO[i]);
        blk[i].ds = ds;
        blk[i].bn1 = {idx, idx+1, idx+2, idx+3}; idx += 4;
        blk[i].bn2 = {idx, idx+1, idx+2, idx+3}; idx += 4;
        if (ds) { blk[i].bnd = {idx, idx+1, idx+2, idx+3}; idx += 4; }
        blk[i].w1 = idx++; blk[i].w2 = idx++;
        if (ds) blk[i].wd = idx++;
    }
    BN bb_bn1 = {idx, idx+1, idx+2, idx+3}; idx += 4;
    int bb_conv1 = idx++;
    int h_b1 = idx++, h_b2 = idx++, h_w1 = idx++, h_w2 = idx++;

    const float* images = (const float*)d_in[0];

    // --- workspace carving (deterministic every call) ---------------------
    char* ws = (char*)d_ws;
    size_t off = 0;
    auto carve = [&](size_t bytes) -> void* {
        void* p = ws + off;
        off = (off + bytes + 255) & ~(size_t)255;
        return p;
    };
    __bf16* imgB = (__bf16*)carve(8ull*3*512*512*2);
    __bf16* bufL = (__bf16*)carve(8ull*64*256*256*2);   // stem conv out
    __bf16* bufX = (__bf16*)carve(8ull*64*128*128*2);   // ping
    __bf16* bufT = (__bf16*)carve(8ull*64*128*128*2);   // intra-block temp
    __bf16* bufY = (__bf16*)carve(8ull*64*128*128*2);   // pong
    __bf16* bufS = (__bf16*)carve(8ull*64*128*128*2);   // downsample branch
    __bf16* feat = (__bf16*)carve(8ull*32*32*32*2);
    float*  hm3  = (float*)carve(8ull*3*32*32*4);
    float*  nmsd = (float*)carve(8ull*1024*4);

    auto packW = [&](int widx, int Co, int Ksrc) -> __bf16* {
        int Kpad = (Ksrc + 31) & ~31;
        __bf16* dst = (__bf16*)carve((size_t)Co * Kpad * sizeof(__bf16));
        int n = Co * Kpad;
        pack_weights<<<(n + 255) / 256, 256, 0, stream>>>(
            (const float*)d_in[widx], dst, Co, Ksrc, Kpad);
        return dst;
    };
    auto bnPrep = [&](BN bn, int C) -> float* {
        float* sb = (float*)carve((size_t)2 * C * 4);
        bn_prep<<<(C + 255) / 256, 256, 0, stream>>>(
            (const float*)d_in[bn.g], (const float*)d_in[bn.b],
            (const float*)d_in[bn.m], (const float*)d_in[bn.v],
            sb, sb + C, C);
        return sb;
    };
    auto convW = [&](const __bf16* pin, const __bf16* wp, const float* sb,
                     const __bf16* res, __bf16* pout,
                     int Ci, int Hi, int Wi, int Co,
                     int stride, int ks, int pad, int relu) {
        int Ho = Hi / stride, Wo = Wi / stride;   // square maps
        int woShift = 31 - __builtin_clz(Wo);
        int Mpix = 8 * Ho * Wo;
        int K = Ci * ks * ks;
        int Kpad = (K + 31) & ~31;
        if (ks == 7) {
            dim3 g(Mpix / 128, Co / 64);
            wmma_conv<64,7><<<g, 256, 0, stream>>>(pin, wp, sb, sb + Co, res, pout,
                Ci, Hi, Wi, Co, woShift, stride, pad, relu, K, Kpad);
        } else if (ks == 3 && (Co % 64 == 0)) {
            dim3 g(Mpix / 128, Co / 64);
            wmma_conv<64,3><<<g, 256, 0, stream>>>(pin, wp, sb, sb + Co, res, pout,
                Ci, Hi, Wi, Co, woShift, stride, pad, relu, K, Kpad);
        } else if (ks == 3) {
            dim3 g(Mpix / 256, Co / 32);
            wmma_conv<32,3><<<g, 256, 0, stream>>>(pin, wp, sb, sb + Co, res, pout,
                Ci, Hi, Wi, Co, woShift, stride, pad, relu, K, Kpad);
        } else {
            dim3 g(Mpix / 128, Co / 64);
            wmma_conv<64,1><<<g, 256, 0, stream>>>(pin, wp, sb, sb + Co, res, pout,
                Ci, Hi, Wi, Co, woShift, stride, pad, relu, K, Kpad);
        }
    };

    // --- stem (WMMA implicit GEMM, K=147 padded to 160) -------------------
    {
        int n = 8 * 3 * 512 * 512;
        f32_to_bf16<<<(n + 255) / 256, 256, 0, stream>>>(images, imgB, n);
    }
    float* sb0 = bnPrep(bb_bn1, 64);
    __bf16* w0p = packW(bb_conv1, 64, 3 * 49);
    convW(imgB, w0p, sb0, nullptr, bufL, 3, 512, 512, 64, 2, 7, 3, 1);
    maxpool3x3s2<<<(8 * 64 * 128 * 128) / 256, 256, 0, stream>>>(bufL, bufX);

    // --- basic blocks -----------------------------------------------------
    __bf16* X = bufX;
    __bf16* Y = bufY;
    int Ci = 64, H = 128;
    for (int i = 0; i < 13; ++i) {
        int Co = chO[i], s = strides[i];
        int Ho = H / s;
        __bf16* w1p = packW(blk[i].w1, Co, Ci * 9);
        __bf16* w2p = packW(blk[i].w2, Co, Co * 9);
        float* sb1 = bnPrep(blk[i].bn1, Co);
        float* sb2 = bnPrep(blk[i].bn2, Co);
        const __bf16* res = X;
        if (blk[i].ds) {
            __bf16* wdp = packW(blk[i].wd, Co, Ci);
            float* sbd = bnPrep(blk[i].bnd, Co);
            convW(X, wdp, sbd, nullptr, bufS, Ci, H, H, Co, s, 1, 0, 0);
            res = bufS;
        }
        convW(X, w1p, sb1, nullptr, bufT, Ci, H, H, Co, s, 3, 1, 1);
        convW(bufT, w2p, sb2, res, Y, Co, Ho, Ho, Co, 1, 3, 1, 1);
        __bf16* tmp = X; X = Y; Y = tmp;
        Ci = Co; H = Ho;
    }

    // --- head -------------------------------------------------------------
    __bf16* hw1p = packW(h_w1, 32, 256 * 9);
    float* hsb = (float*)carve(2 * 32 * 4);
    bias_prep<<<1, 64, 0, stream>>>((const float*)d_in[h_b1], hsb, hsb + 32, 32);
    convW(X, hw1p, hsb, nullptr, feat, 256, 32, 32, 32, 1, 3, 1, 1);
    conv1x1_head<<<(8 * 3 * 1024 + 255) / 256, 256, 0, stream>>>(
        feat, (const float*)d_in[h_w2], (const float*)d_in[h_b2], hm3);

    // --- post-processing --------------------------------------------------
    softmax_nms<<<8, 256, 0, stream>>>(hm3, nmsd);
    topk_gather<<<8, 256, 0, stream>>>(nmsd, hm3, (float*)d_out);
}
